// AttentiveStatPooling_block_36412732735577
// MI455X (gfx1250) — compile-verified, matrix operand order checked by reading, not run on hardware
//
#include <hip/hip_runtime.h>
#include <hip/hip_bf16.h>

// ---------- WMMA types (CDNA5 / gfx1250, wave32) ----------
typedef __attribute__((ext_vector_type(16))) __bf16 v16bf;
typedef __attribute__((ext_vector_type(8)))  float  v8f;
typedef unsigned int u32x4 __attribute__((ext_vector_type(4)));
typedef int          i32x8 __attribute__((ext_vector_type(8)));
typedef int          i32x4 __attribute__((ext_vector_type(4)));

union Frag { v16bf v; uint4 q[2]; };

#define BN_EPS 1e-5f
#define CLAMPV 1e-4f

static __device__ __forceinline__ float wave16_max(float v) {
  #pragma unroll
  for (int m = 1; m < 16; m <<= 1) v = fmaxf(v, __shfl_xor(v, m, 32));
  return v;
}
static __device__ __forceinline__ float wave16_sum(float v) {
  #pragma unroll
  for (int m = 1; m < 16; m <<= 1) v += __shfl_xor(v, m, 32);
  return v;
}
// fast tanh: (e^{2x}-1)/(e^{2x}+1), input clamped (tanh saturated anyway)
static __device__ __forceinline__ float fast_tanh(float x) {
  x = fminf(fmaxf(x, -10.f), 10.f);
  float e = __expf(2.f * x);
  return (e - 1.f) / (e + 1.f);
}

// ---------- Kernel 1: per-(b,c) mean/std over T=1000 ----------
__global__ __launch_bounds__(256) void stats_kernel(
    const float* __restrict__ x, float* __restrict__ meanp, float* __restrict__ stdp) {
  int row  = blockIdx.x * 8 + (threadIdx.x >> 5);   // one wave per (b,c) row
  int lane = threadIdx.x & 31;
  if (row >= 32 * 1536) return;
  const float* xr = x + (size_t)row * 1000;
  float s = 0.f, s2 = 0.f;
  for (int t = lane; t < 1000; t += 32) { float v = xr[t]; s += v; s2 += v * v; }
  #pragma unroll
  for (int m = 1; m < 32; m <<= 1) { s += __shfl_xor(s, m, 32); s2 += __shfl_xor(s2, m, 32); }
  if (lane == 0) {
    float mean = s * (1.f / 1000.f);
    float var  = (s2 - 1000.f * mean * mean) * (1.f / 999.f);   // ddof=1
    meanp[row] = mean;
    stdp[row]  = sqrtf(fmaxf(var, CLAMPV));
  }
}

// ---------- Kernel 2: repack Wx/W2 to bf16, fold BN params ----------
__global__ __launch_bounds__(256) void prep_kernel(
    const float* __restrict__ W1, const float* __restrict__ W2,
    const float* __restrict__ g1, const float* __restrict__ beta1,
    const float* __restrict__ rm1, const float* __restrict__ rv1,
    const float* __restrict__ g2, const float* __restrict__ beta2,
    const float* __restrict__ rm2, const float* __restrict__ rv2,
    __bf16* __restrict__ Wxb, __bf16* __restrict__ W2b,
    float* __restrict__ inv1, float* __restrict__ sh1,
    float* __restrict__ inv2, float* __restrict__ sh2) {
  int i = blockIdx.x * 256 + threadIdx.x;
  const int NW = 128 * 1536;
  if (i < NW) {
    int o = i / 1536, c = i - o * 1536;
    Wxb[i] = (__bf16)W1[(size_t)o * 4608 + c];   // W1[:, :C]
    W2b[i] = (__bf16)W2[i];                      // (C, A) row-major
  }
  if (i < 128) {
    float inv = g1[i] * rsqrtf(rv1[i] + BN_EPS);
    inv1[i] = inv; sh1[i] = beta1[i] - rm1[i] * inv;
  } else if (i < 128 + 1536) {
    int c = i - 128;
    float inv = g2[c] * rsqrtf(rv2[c] + BN_EPS);
    inv2[c] = inv; sh2[c] = beta2[c] - rm2[c] * inv;
  }
}

// ---------- Kernel 3: bias1[b,o] = b1[o] + mean@Wm^T + std@Ws^T ----------
__global__ __launch_bounds__(256) void bias1_kernel(
    const float* __restrict__ W1, const float* __restrict__ b1,
    const float* __restrict__ meanp, const float* __restrict__ stdp,
    float* __restrict__ bias1) {
  int id   = blockIdx.x * 8 + (threadIdx.x >> 5);  // one wave per (b,o)
  int lane = threadIdx.x & 31;
  if (id >= 32 * 128) return;
  int b = id >> 7, o = id & 127;
  const float* wm = W1 + (size_t)o * 4608 + 1536;  // Wm row
  const float* ws = W1 + (size_t)o * 4608 + 3072;  // Ws row
  const float* mr = meanp + b * 1536;
  const float* sr = stdp  + b * 1536;
  float s = 0.f;
  for (int c = lane; c < 1536; c += 32) s += mr[c] * wm[c] + sr[c] * ws[c];
  #pragma unroll
  for (int m = 1; m < 32; m <<= 1) s += __shfl_xor(s, m, 32);
  if (lane == 0) bias1[id] = b1[o] + s;
}

// ---------- Kernel 4: GEMM1 (bf16 WMMA) + bias/relu/BN/tanh -> a_t (bf16) ---
// grid (63, 32); block 256 = 8 waves; wave w computes o-tile w (16 rows).
// Output layout a_t[b][t][o] (Tp=1008 rows of 128 bf16) so each lane packs
// its 8 consecutive-o values into one 16B store, and the per-batch slab is a
// clean 2D tile for the Tensor Data Mover in kernel 5.
#define XROW 1544   // padded k-row stride (bf16 elems) -> bank-conflict free
__global__ __launch_bounds__(256) void gemm1_kernel(
    const float* __restrict__ x, const __bf16* __restrict__ Wxb,
    const float* __restrict__ bias1, const float* __restrict__ inv1,
    const float* __restrict__ sh1, __bf16* __restrict__ a_t) {
  extern __shared__ char smem[];
  __bf16* lds_x = (__bf16*)smem;                 // [16][XROW] : t-major rows over k
  const int b  = blockIdx.y;
  const int t0 = blockIdx.x * 16;
  const int tid = threadIdx.x;

  // stage x[b, :, t0:t0+16] -> bf16 LDS, transposed to [t][k]
  for (int idx = tid; idx < 1536 * 4; idx += 256) {
    int c = idx >> 2, t4 = (idx & 3) * 4;
    const float* xp = x + ((size_t)(b * 1536 + c)) * 1000 + t0 + t4;
    float v0 = 0.f, v1 = 0.f, v2 = 0.f, v3 = 0.f;
    if (t0 + t4 + 3 < 1000) {
      float4 v = *(const float4*)xp; v0 = v.x; v1 = v.y; v2 = v.z; v3 = v.w;
    } else {
      if (t0 + t4 + 0 < 1000) v0 = xp[0];
      if (t0 + t4 + 1 < 1000) v1 = xp[1];
      if (t0 + t4 + 2 < 1000) v2 = xp[2];
      if (t0 + t4 + 3 < 1000) v3 = xp[3];
    }
    lds_x[(t4 + 0) * XROW + c] = (__bf16)v0;
    lds_x[(t4 + 1) * XROW + c] = (__bf16)v1;
    lds_x[(t4 + 2) * XROW + c] = (__bf16)v2;
    lds_x[(t4 + 3) * XROW + c] = (__bf16)v3;
  }
  __syncthreads();

  const int wave = tid >> 5, lane = tid & 31;
  const int half = lane >> 4, n = lane & 15;
  const int otile = wave;
  const __bf16* wrow = Wxb + (size_t)(otile * 16 + n) * 1536;  // A: lane holds row m=n
  const __bf16* lrow = lds_x + n * XROW;                       // B: lane holds col t=n

  v8f acc = {};
  for (int ks = 0; ks < 48; ++ks) {               // K = 1536 = 48 * 32
    Frag af, bf;
    af.q[0] = *(const uint4*)(wrow + ks * 32 + half * 8);
    af.q[1] = *(const uint4*)(wrow + ks * 32 + 16 + half * 8);
    bf.q[0] = *(const uint4*)(lrow + ks * 32 + half * 8);
    bf.q[1] = *(const uint4*)(lrow + ks * 32 + 16 + half * 8);
    acc = __builtin_amdgcn_wmma_f32_16x16x32_bf16(
        false, af.v, false, bf.v, (short)0, acc, false, false);
  }

  // epilogue: +bias1, relu, BN1, fast tanh -> packed 16B store of 8 o-values
  uint4 pk;
  __bf16* pe = (__bf16*)&pk;
  #pragma unroll
  for (int r = 0; r < 8; ++r) {
    int o = otile * 16 + half * 8 + r;            // C/D row layout (consecutive o)
    float v = acc[r] + bias1[b * 128 + o];
    v = fmaxf(v, 0.f);
    v = v * inv1[o] + sh1[o];
    pe[r] = (__bf16)fast_tanh(v);
  }
  *(uint4*)(a_t + ((size_t)(b * 1008 + t0 + n)) * 128 + otile * 16 + half * 8) = pk;
}

// ---------- Kernel 5: GEMM2 (bf16 WMMA) + relu/BN + online softmax + mu/sg --
// grid (6, 32); block 512 = 16 waves. Whole a_t[b] (1008x128 bf16) staged into
// LDS with bank padding (row stride 136 -> 274 KB) by the Tensor Data Mover:
// pad_interval=5 (every 64 DWORDs = one 128-bf16 row), pad_amount=3 (4 DWORDs).
#define AROW 136
__global__ __launch_bounds__(512) void gemm2_pool_kernel(
    const __bf16* __restrict__ a_t, const __bf16* __restrict__ W2b,
    const float* __restrict__ x, const float* __restrict__ b2,
    const float* __restrict__ inv2, const float* __restrict__ sh2,
    float* __restrict__ out) {
  extern __shared__ char smem[];
  __bf16* lds_a = (__bf16*)smem;                 // [1008][AROW] : t-major over k=o
  const int b = blockIdx.y;
  const int tid = threadIdx.x;
  const int wave = tid >> 5, lane = tid & 31;

#if __has_builtin(__builtin_amdgcn_tensor_load_to_lds) && __has_builtin(__builtin_amdgcn_s_wait_tensorcnt)
  if (wave == 0) {
    unsigned long long ga =
        (unsigned long long)(uintptr_t)(a_t + (size_t)b * 1008 * 128);
    // D# group0: count=1 | lds_addr=0 | global_addr | type=2 ("image")
    u32x4 g0 = { 1u, 0u, (unsigned)ga,
                 (unsigned)(((ga >> 32) & 0x1FFFFFFull) | (2u << 30)) };
    // D# group1: data_size=2B, pad_enable, pad_interval=5 (64 dw), pad_amount=3 (4 dw)
    //            tensor_dim0=128, tensor_dim1=1008, tile_dim0=128, tile_dim1=1008,
    //            tensor_dim0_stride=128
    i32x8 g1 = { (int)((1u << 16) | (1u << 20) | (5u << 22) | (3u << 25)),
                 (int)(128u << 16),     // tensor_dim0[15:0] @ bit48
                 (int)(1008u << 16),    // tensor_dim1[15:0] @ bit80
                 (int)(128u << 16),     // tile_dim0 @ bits127:112
                 1008,                  // tile_dim1
                 128,                   // tensor_dim0_stride
                 0, 0 };
    i32x4 gz4 = { 0, 0, 0, 0 };
    i32x8 gz8 = { 0, 0, 0, 0, 0, 0, 0, 0 };
    __builtin_amdgcn_tensor_load_to_lds(g0, g1, gz4, gz4, gz8, 0);
    __builtin_amdgcn_s_wait_tensorcnt(0);
  }
#else
  // fallback: manual 16B-chunk staging with bank padding
  for (int idx = tid; idx < 1008 * 16; idx += 512) {
    int t = idx >> 4, oc = (idx & 15) * 8;
    uint4 v = *(const uint4*)(a_t + ((size_t)(b * 1008 + t)) * 128 + oc);
    *(uint4*)(lds_a + t * AROW + oc) = v;
  }
#endif
  __syncthreads();

  const int half = lane >> 4, n = lane & 15;
  const int c0 = (blockIdx.x * 16 + wave) * 16;

  // W2 A-fragments (16 rows c x 128 k), register-resident for whole t loop
  Frag w2f[4];
  const __bf16* w2row = W2b + (size_t)(c0 + n) * 128;
  #pragma unroll
  for (int ks = 0; ks < 4; ++ks) {
    w2f[ks].q[0] = *(const uint4*)(w2row + ks * 32 + half * 8);
    w2f[ks].q[1] = *(const uint4*)(w2row + ks * 32 + 16 + half * 8);
  }
  float b2r[8], ivr[8], shr[8];
  #pragma unroll
  for (int r = 0; r < 8; ++r) {
    int c = c0 + half * 8 + r;
    b2r[r] = b2[c]; ivr[r] = inv2[c]; shr[r] = sh2[c];
  }

  // per-lane online softmax state (no cross-lane traffic in the hot loop)
  float mrun[8], S0[8], S1[8], S2[8];
  #pragma unroll
  for (int r = 0; r < 8; ++r) { mrun[r] = -3.0e38f; S0[r] = S1[r] = S2[r] = 0.f; }

  for (int tt = 0; tt < 63; ++tt) {               // T=1000 -> 63 tiles (pad masked)
    const int t0 = tt * 16;
    v8f acc = {};
    const __bf16* arow = lds_a + (t0 + n) * AROW; // B: lane holds col t
    #pragma unroll
    for (int ks = 0; ks < 4; ++ks) {
      Frag af;
      af.q[0] = *(const uint4*)(arow + ks * 32 + half * 8);
      af.q[1] = *(const uint4*)(arow + ks * 32 + 16 + half * 8);
      acc = __builtin_amdgcn_wmma_f32_16x16x32_bf16(
          false, w2f[ks].v, false, af.v, (short)0, acc, false, false);
    }
    const int t = t0 + n;
    if (t < 1000) {
      #pragma unroll
      for (int r = 0; r < 8; ++r) {
        float h = fmaxf(acc[r] + b2r[r], 0.f);    // relu
        h = h * ivr[r] + shr[r];                  // BN2
        int c = c0 + half * 8 + r;
        float xv = x[((size_t)(b * 1536 + c)) * 1000 + t];
        float mnew = fmaxf(mrun[r], h);
        float sc = __expf(mrun[r] - mnew);        // per-lane online rescale
        float w  = __expf(h - mnew);
        S0[r] = S0[r] * sc + w;
        S1[r] = S1[r] * sc + w * xv;
        S2[r] = S2[r] * sc + w * xv * xv;
        mrun[r] = mnew;
      }
    }
  }

  // combine lanes: rescale each lane's partials to the row max, then sum
  #pragma unroll
  for (int r = 0; r < 8; ++r) {
    float M = wave16_max(mrun[r]);
    float f = __expf(mrun[r] - M);
    float s0 = wave16_sum(S0[r] * f);
    float s1 = wave16_sum(S1[r] * f);
    float s2 = wave16_sum(S2[r] * f);
    if (n == 0) {
      int c = c0 + half * 8 + r;
      float mu = s1 / s0;
      float sg = sqrtf(fmaxf(s2 / s0 - mu * mu, CLAMPV));
      out[b * 3072 + c]        = mu;              // concat([mu, sg], axis=1)
      out[b * 3072 + 1536 + c] = sg;
    }
  }
}

// ---------- host launcher ----------
extern "C" void kernel_launch(void* const* d_in, const int* in_sizes, int n_in,
                              void* d_out, int out_size, void* d_ws, size_t ws_size,
                              hipStream_t stream) {
  (void)in_sizes; (void)n_in; (void)out_size; (void)ws_size;
  const float* x    = (const float*)d_in[0];
  const float* W1   = (const float*)d_in[1];
  const float* b1   = (const float*)d_in[2];
  const float* g1   = (const float*)d_in[3];
  const float* bt1  = (const float*)d_in[4];
  const float* rm1  = (const float*)d_in[5];
  const float* rv1  = (const float*)d_in[6];
  const float* W2   = (const float*)d_in[7];
  const float* b2   = (const float*)d_in[8];
  const float* g2   = (const float*)d_in[9];
  const float* bt2  = (const float*)d_in[10];
  const float* rm2  = (const float*)d_in[11];
  const float* rv2  = (const float*)d_in[12];
  float* out = (float*)d_out;

  char* ws = (char*)d_ws;
  size_t off = 0;
  auto carve = [&](size_t bytes) -> char* {
    char* p = ws + off;
    off += (bytes + 255) & ~(size_t)255;
    return p;
  };
  float*  meanp = (float*)carve(32 * 1536 * 4);
  float*  stdp  = (float*)carve(32 * 1536 * 4);
  float*  bias1 = (float*)carve(32 * 128 * 4);
  float*  inv1  = (float*)carve(128 * 4);
  float*  sh1   = (float*)carve(128 * 4);
  float*  inv2  = (float*)carve(1536 * 4);
  float*  sh2   = (float*)carve(1536 * 4);
  __bf16* Wxb   = (__bf16*)carve((size_t)128 * 1536 * 2);
  __bf16* W2b   = (__bf16*)carve((size_t)1536 * 128 * 2);
  __bf16* a_t   = (__bf16*)carve((size_t)32 * 1008 * 128 * 2);

  stats_kernel<<<dim3(32 * 1536 / 8), dim3(256), 0, stream>>>(x, meanp, stdp);
  prep_kernel<<<dim3((128 * 1536 + 255) / 256), dim3(256), 0, stream>>>(
      W1, W2, g1, bt1, rm1, rv1, g2, bt2, rm2, rv2, Wxb, W2b, inv1, sh1, inv2, sh2);
  bias1_kernel<<<dim3(32 * 128 / 8), dim3(256), 0, stream>>>(W1, b1, meanp, stdp, bias1);
  gemm1_kernel<<<dim3(63, 32), dim3(256), 16 * XROW * 2, stream>>>(
      x, Wxb, bias1, inv1, sh1, a_t);
  gemm2_pool_kernel<<<dim3(6, 32), dim3(512), 1008 * AROW * 2, stream>>>(
      a_t, W2b, x, b2, inv2, sh2, out);
}
